// Net_16999480558201
// MI455X (gfx1250) — matrix operand order repaired and checked by reading, not compiled, not run on hardware
//
#include <hip/hip_runtime.h>

typedef __attribute__((ext_vector_type(16))) _Float16 v16h;
typedef __attribute__((ext_vector_type(8)))  _Float16 v8h;
typedef __attribute__((ext_vector_type(8)))  float    v8f;

union V16U { v16h v; v8h h[2]; };

static __device__ __forceinline__ v8f wmma16(v16h a, v16h b, v8f c) {
  // D = A(16x32 f16) * B(32x16 f16) + C(16x16 f32)
  return __builtin_amdgcn_wmma_f32_16x16x32_f16(false, a, false, b, (short)0, c,
                                                false, false);
}

// ---------------- prep kernels ----------------

// b0c[n] = b0[n] + sum_k x_graph[k] * W0[k][n], k = 0..255 (constant part folded)
__global__ void k_prep_b0(const float* __restrict__ xg, const float* __restrict__ W0,
                          const float* __restrict__ b0, float* __restrict__ b0c) {
  int nn = threadIdx.x; // 128 threads
  float acc = b0[nn];
  for (int k = 0; k < 256; ++k) acc += xg[k] * W0[k * 128 + nn];
  b0c[nn] = acc;
}

// W0t[n][k] = (f16) W0[256+k][n]   (n<128, k<256)
__global__ void k_w0t(const float* __restrict__ W0, _Float16* __restrict__ W0t) {
  int idx = blockIdx.x * 256 + threadIdx.x; // 32768 total
  int nn = idx >> 8, k = idx & 255;
  W0t[nn * 256 + k] = (_Float16)W0[(256 + k) * 128 + nn];
}

// W1t[n][k] = (f16) W1[k][n]   (n<128, k<128)
__global__ void k_w1t(const float* __restrict__ W1, _Float16* __restrict__ W1t) {
  int idx = blockIdx.x * 256 + threadIdx.x; // 16384 total
  int nn = idx >> 7, k = idx & 127;
  W1t[nn * 128 + k] = (_Float16)W1[k * 128 + nn];
}

__global__ void k_cvt(const float* __restrict__ src, _Float16* __restrict__ dst, int count) {
  int i = blockIdx.x * blockDim.x + threadIdx.x;
  if (i < count) dst[i] = (_Float16)src[i];
}

// ---------------- main MLP kernel ----------------
// 128 threads = 4 waves; each wave computes 32 rows (two 16-row A tiles sharing
// every B fragment -> 1 ds_load_b128 per WMMA instead of 2). Block covers 128 rows.
// Weight K-slices (128 cols x 32 K, f16) are double-buffered through LDS:
// one global copy per block instead of one per wave.
#define WSL_STRIDE 40  // halves per slice row (padded from 32 to dodge bank conflicts)

__global__ __launch_bounds__(128) void k_mlp(
    const int* __restrict__ m_ids, const int* __restrict__ job_idx,
    const _Float16* __restrict__ xm16, const _Float16* __restrict__ xj16,
    const _Float16* __restrict__ W0t,   // [128][256]
    const _Float16* __restrict__ W1t,   // [128][128]
    const float* __restrict__ b0c,      // [128] (x_graph folded in)
    const float* __restrict__ b1,       // [128]
    const float* __restrict__ W2,       // [128]
    const float* __restrict__ b2,       // [1]
    float* __restrict__ scores, int n) {
  __shared__ __align__(16) _Float16 lds_h0[4][32][136];          // per-wave 32x128 (+pad)
  __shared__ __align__(16) _Float16 wslice[2][128 * WSL_STRIDE]; // double-buffered K-slice

  const int tid  = threadIdx.x;
  const int wave = tid >> 5;
  const int lane = tid & 31;
  const int col  = lane & 15;   // matrix column for B/C/D, row for A
  const int hs   = lane >> 4;   // half-wave select
  const int row_base = blockIdx.x * 128 + wave * 32;

  int myrow0 = row_base + col;
  int myrow1 = row_base + 16 + col;
  int r0 = myrow0 < n ? myrow0 : 0;
  int r1 = myrow1 < n ? myrow1 : 0;
  const _Float16* arow_m0 = xm16 + (size_t)m_ids[r0]  * 128; // tile0 feats K 0..127
  const _Float16* arow_j0 = xj16 + (size_t)job_idx[r0] * 128; // tile0 feats K 128..255
  const _Float16* arow_m1 = xm16 + (size_t)m_ids[r1]  * 128; // tile1
  const _Float16* arow_j1 = xj16 + (size_t)job_idx[r1] * 128;

  float b0v[8], b1v[8], w2v[8];
#pragma unroll
  for (int ct = 0; ct < 8; ++ct) {
    b0v[ct] = b0c[ct * 16 + col];
    b1v[ct] = b1[ct * 16 + col];
    w2v[ct] = W2[ct * 16 + col];
  }

  // cooperative copy of one 128x32 f16 K-slice into wslice[bufsel]
  // 128 threads: each copies one 64 B slice row (4x b128 load + 4x ds_store_b128)
  auto stage = [&](const _Float16* __restrict__ Wt, int rowStride, int kk, int bufsel) {
    const _Float16* s = Wt + (size_t)tid * rowStride + kk * 32;
    v8h x0 = *(const v8h*)(s);
    v8h x1 = *(const v8h*)(s + 8);
    v8h x2 = *(const v8h*)(s + 16);
    v8h x3 = *(const v8h*)(s + 24);
    _Float16* d = &wslice[bufsel][tid * WSL_STRIDE];
    *(v8h*)(d)      = x0;
    *(v8h*)(d + 8)  = x1;
    *(v8h*)(d + 16) = x2;
    *(v8h*)(d + 24) = x3;
  };

  // ---- layer 0: [32x256] @ [256x128], K in 8 steps of 32 ----
  v8f acc[2][8] = {};
  stage(W0t, 256, 0, 0);
  __syncthreads();
#pragma unroll
  for (int kk = 0; kk < 8; ++kk) {
    if (kk < 7) stage(W0t, 256, kk + 1, (kk + 1) & 1); // prefetch next slice
    const _Float16* ar0 = (kk < 4) ? arow_m0 : arow_j0;
    const _Float16* ar1 = (kk < 4) ? arow_m1 : arow_j1;
    const int kb = (kk & 3) * 32;
    // ISA 16-bit A layout: halves 0..7 -> K=kb+8*hs.., halves 8..15 -> K=kb+16+8*hs..
    V16U a0, a1;
    a0.h[0] = *(const v8h*)(ar0 + kb + 8 * hs);
    a0.h[1] = *(const v8h*)(ar0 + kb + 16 + 8 * hs);
    a1.h[0] = *(const v8h*)(ar1 + kb + 8 * hs);
    a1.h[1] = *(const v8h*)(ar1 + kb + 16 + 8 * hs);
    const _Float16* wcur = &wslice[kk & 1][0];
#pragma unroll
    for (int ct = 0; ct < 8; ++ct) {
      // B layout: lane col = N, halves 0..7 -> K=8*hs.., halves 8..15 -> K=16+8*hs..
      const _Float16* bp = wcur + (ct * 16 + col) * WSL_STRIDE + 8 * hs;
      V16U b;
      b.h[0] = *(const v8h*)(bp);
      b.h[1] = *(const v8h*)(bp + 16);
      acc[0][ct] = wmma16(a0.v, b.v, acc[0][ct]);
      acc[1][ct] = wmma16(a1.v, b.v, acc[1][ct]);
    }
    __syncthreads();
  }

  // bias + ReLU, re-layout D->A via wave-private LDS tile
#pragma unroll
  for (int t = 0; t < 2; ++t) {
#pragma unroll
    for (int ct = 0; ct < 8; ++ct) {
#pragma unroll
      for (int r = 0; r < 8; ++r) {
        float v = acc[t][ct][r] + b0v[ct];
        v = v > 0.0f ? v : 0.0f;
        lds_h0[wave][t * 16 + r + 8 * hs][ct * 16 + col] = (_Float16)v;
      }
    }
  }
  __syncthreads();

  // ---- layer 1: [32x128] @ [128x128], K in 4 steps of 32 ----
  v8f acc2[2][8] = {};
  const _Float16* hrow0 = &lds_h0[wave][col][0];
  const _Float16* hrow1 = &lds_h0[wave][16 + col][0];
  stage(W1t, 128, 0, 0);
  __syncthreads();
#pragma unroll
  for (int kk = 0; kk < 4; ++kk) {
    if (kk < 3) stage(W1t, 128, kk + 1, (kk + 1) & 1);
    const int kb = kk * 32;
    V16U a0, a1;
    a0.h[0] = *(const v8h*)(hrow0 + kb + 8 * hs);
    a0.h[1] = *(const v8h*)(hrow0 + kb + 16 + 8 * hs);
    a1.h[0] = *(const v8h*)(hrow1 + kb + 8 * hs);
    a1.h[1] = *(const v8h*)(hrow1 + kb + 16 + 8 * hs);
    const _Float16* wcur = &wslice[kk & 1][0];
#pragma unroll
    for (int ct = 0; ct < 8; ++ct) {
      const _Float16* bp = wcur + (ct * 16 + col) * WSL_STRIDE + 8 * hs;
      V16U b;
      b.h[0] = *(const v8h*)(bp);
      b.h[1] = *(const v8h*)(bp + 16);
      acc2[0][ct] = wmma16(a0.v, b.v, acc2[0][ct]);
      acc2[1][ct] = wmma16(a1.v, b.v, acc2[1][ct]);
    }
    __syncthreads();
  }

  // ---- layer 2: ReLU then dot with W2, straight off the accumulators ----
  float p[2][8];
#pragma unroll
  for (int t = 0; t < 2; ++t)
#pragma unroll
    for (int r = 0; r < 8; ++r) p[t][r] = 0.0f;
#pragma unroll
  for (int t = 0; t < 2; ++t) {
#pragma unroll
    for (int ct = 0; ct < 8; ++ct) {
#pragma unroll
      for (int r = 0; r < 8; ++r) {
        float h = acc2[t][ct][r] + b1v[ct];
        h = h > 0.0f ? h : 0.0f;
        p[t][r] += h * w2v[ct];
      }
    }
  }
  // reduce over the 16 lanes of each half-wave (columns 0..15)
#pragma unroll
  for (int t = 0; t < 2; ++t) {
#pragma unroll
    for (int r = 0; r < 8; ++r) {
#pragma unroll
      for (int mk = 1; mk < 16; mk <<= 1) p[t][r] += __shfl_xor(p[t][r], mk, 32);
    }
  }
  if (col < 8) {
#pragma unroll
    for (int t = 0; t < 2; ++t) {
      int rw = row_base + t * 16 + 8 * hs + col;
      if (rw < n) scores[rw] = p[t][col] + b2[0];
    }
  }
}

// ---------------- deterministic reductions ----------------

static __device__ __forceinline__ unsigned orderable(float f) {
  unsigned u = __float_as_uint(f);
  return (u & 0x80000000u) ? ~u : (u | 0x80000000u);
}

__global__ __launch_bounds__(256) void k_block_max(const float* __restrict__ scores,
                                                   int n, unsigned long long* __restrict__ bkeys) {
  __shared__ unsigned long long sk[256];
  unsigned long long best = 0ull;
  for (int i = blockIdx.x * blockDim.x + threadIdx.x; i < n; i += gridDim.x * blockDim.x) {
    unsigned long long key = ((unsigned long long)orderable(scores[i]) << 32) |
                             (unsigned long long)(0xFFFFFFFFu - (unsigned)i); // ties -> lowest idx
    best = key > best ? key : best;
  }
  sk[threadIdx.x] = best;
  __syncthreads();
  for (int s = 128; s > 0; s >>= 1) {
    if (threadIdx.x < s && sk[threadIdx.x + s] > sk[threadIdx.x]) sk[threadIdx.x] = sk[threadIdx.x + s];
    __syncthreads();
  }
  if (threadIdx.x == 0) bkeys[blockIdx.x] = sk[0];
}

__global__ void k_max_final(const unsigned long long* __restrict__ bkeys, int nkeys,
                            float* __restrict__ res) {
  __shared__ unsigned long long sk[256];
  sk[threadIdx.x] = (threadIdx.x < nkeys) ? bkeys[threadIdx.x] : 0ull;
  __syncthreads();
  for (int s = 128; s > 0; s >>= 1) {
    if (threadIdx.x < s && sk[threadIdx.x + s] > sk[threadIdx.x]) sk[threadIdx.x] = sk[threadIdx.x + s];
    __syncthreads();
  }
  if (threadIdx.x == 0) {
    unsigned long long k = sk[0];
    unsigned um = (unsigned)(k >> 32);
    unsigned idx = 0xFFFFFFFFu - (unsigned)(k & 0xFFFFFFFFu);
    unsigned uo = (um & 0x80000000u) ? (um ^ 0x80000000u) : ~um;
    res[0] = __uint_as_float(uo); // max score
    res[1] = (float)idx;          // argmax
  }
}

__global__ __launch_bounds__(256) void k_partials(const float* __restrict__ scores, int n,
                                                  const float* __restrict__ res,
                                                  float* __restrict__ partials) {
  __shared__ float s0s[256], s1s[256];
  const float maxs = res[0];
  float s0 = 0.0f, s1 = 0.0f;
  for (int i = blockIdx.x * blockDim.x + threadIdx.x; i < n; i += gridDim.x * blockDim.x) {
    float d = scores[i] - maxs;
    float e = expf(d);
    s0 += e;
    s1 += d * e;
  }
  s0s[threadIdx.x] = s0; s1s[threadIdx.x] = s1;
  __syncthreads();
  for (int s = 128; s > 0; s >>= 1) {
    if (threadIdx.x < s) { s0s[threadIdx.x] += s0s[threadIdx.x + s]; s1s[threadIdx.x] += s1s[threadIdx.x + s]; }
    __syncthreads();
  }
  if (threadIdx.x == 0) { partials[2 * blockIdx.x] = s0s[0]; partials[2 * blockIdx.x + 1] = s1s[0]; }
}

__global__ void k_finalize(const float* __restrict__ partials, int nparts,
                           const float* __restrict__ res, float* __restrict__ out) {
  __shared__ float s0s[256], s1s[256];
  float s0 = 0.0f, s1 = 0.0f;
  if (threadIdx.x < nparts) { s0 = partials[2 * threadIdx.x]; s1 = partials[2 * threadIdx.x + 1]; }
  s0s[threadIdx.x] = s0; s1s[threadIdx.x] = s1;
  __syncthreads();
  for (int s = 128; s > 0; s >>= 1) {
    if (threadIdx.x < s) { s0s[threadIdx.x] += s0s[threadIdx.x + s]; s1s[threadIdx.x] += s1s[threadIdx.x + s]; }
    __syncthreads();
  }
  if (threadIdx.x == 0) {
    float S0 = s0s[0], S1 = s1s[0];
    float logZ = logf(S0);
    out[0] = res[1];          // argmax index
    out[1] = 1.0f / S0;       // probs[idx] = exp(s_idx - max)/S0, s_idx == max
    out[2] = -logZ;           // logp[idx]
    out[3] = logZ - S1 / S0;  // entropy
  }
}

// ---------------- launcher ----------------

extern "C" void kernel_launch(void* const* d_in, const int* in_sizes, int n_in,
                              void* d_out, int out_size, void* d_ws, size_t ws_size,
                              hipStream_t stream) {
  (void)n_in; (void)out_size; (void)ws_size;
  const float* xg   = (const float*)d_in[0];
  const float* xm   = (const float*)d_in[1];
  const float* xjob = (const float*)d_in[2];
  const int*   mids = (const int*)d_in[3];
  const int*   jidx = (const int*)d_in[4];
  const float* W0   = (const float*)d_in[5];
  const float* b0   = (const float*)d_in[6];
  const float* W1   = (const float*)d_in[7];
  const float* b1   = (const float*)d_in[8];
  const float* W2   = (const float*)d_in[9];
  const float* b2   = (const float*)d_in[10];

  const int n     = in_sizes[3];
  const int Mrows = in_sizes[1] / 128;
  const int Jrows = in_sizes[2] / 128;

  char* ws = (char*)d_ws;
  size_t off = 0;
  auto wsalloc = [&](size_t bytes) -> void* {
    void* p = ws + off;
    off = (off + bytes + 255) & ~(size_t)255;
    return p;
  };
  float*              scores   = (float*)wsalloc((size_t)n * sizeof(float));
  _Float16*           W0t      = (_Float16*)wsalloc(128 * 256 * sizeof(_Float16));
  _Float16*           W1t      = (_Float16*)wsalloc(128 * 128 * sizeof(_Float16));
  _Float16*           xm16     = (_Float16*)wsalloc((size_t)Mrows * 128 * sizeof(_Float16));
  _Float16*           xj16     = (_Float16*)wsalloc((size_t)Jrows * 128 * sizeof(_Float16));
  float*              b0c      = (float*)wsalloc(128 * sizeof(float));
  unsigned long long* bkeys    = (unsigned long long*)wsalloc(256 * sizeof(unsigned long long));
  float*              res      = (float*)wsalloc(256);
  float*              partials = (float*)wsalloc(256 * 2 * sizeof(float));

  k_prep_b0<<<1, 128, 0, stream>>>(xg, W0, b0, b0c);
  k_w0t<<<128, 256, 0, stream>>>(W0, W0t);
  k_w1t<<<64, 256, 0, stream>>>(W1, W1t);
  const int cm = Mrows * 128, cj = Jrows * 128;
  k_cvt<<<(cm + 255) / 256, 256, 0, stream>>>(xm, xm16, cm);
  k_cvt<<<(cj + 255) / 256, 256, 0, stream>>>(xjob, xj16, cj);

  k_mlp<<<(n + 127) / 128, 128, 0, stream>>>(mids, jidx, xm16, xj16, W0t, W1t,
                                             b0c, b1, W2, b2, scores, n);

  k_block_max<<<256, 256, 0, stream>>>(scores, n, bkeys);
  k_max_final<<<1, 256, 0, stream>>>(bkeys, 256, res);
  k_partials<<<256, 256, 0, stream>>>(scores, n, res, partials);
  k_finalize<<<1, 256, 0, stream>>>(partials, 256, res, (float*)d_out);
}
